// CATSI_37203006717985
// MI455X (gfx1250) — compile-verified, matrix operand order checked by reading, not run on hardware
//
#include <hip/hip_runtime.h>
#include <hip/hip_bf16.h>

#define BB 256
#define TSTEPS 512
#define VV 32
#define BTROWS (BB * TSTEPS)  // 131072

typedef __attribute__((ext_vector_type(16))) _Float16 v16h;
typedef __attribute__((ext_vector_type(8)))  _Float16 v8h;
typedef __attribute__((ext_vector_type(8)))  float    v8f;

__device__ __forceinline__ float sigf(float x) { return 1.0f / (1.0f + __expf(-x)); }

// Load a 16x32 f16 fragment (A-layout; also used for B via W[N][K] rows).
// lanes 0-15: row=row0+lane, cols k0+[0..7] and k0+[16..23]
// lanes 16-31: row=row0+lane-16, cols k0+[8..15] and k0+[24..31]
__device__ __forceinline__ v16h frag_load(const _Float16* p0, int ld) {
  int lane = threadIdx.x & 31;
  const _Float16* p = p0 + (lane & 15) * ld + ((lane >> 4) << 3);
  v8h lo = *(const v8h*)p;
  v8h hi = *(const v8h*)(p + 16);
  v16h r;
#pragma unroll
  for (int i = 0; i < 8; ++i) { r[i] = lo[i]; r[i + 8] = hi[i]; }
  return r;
}

// Fire-and-forget async copy of 16 bytes global -> LDS (per lane addressing).
__device__ __forceinline__ void async_b128(unsigned lds_off, const void* gaddr) {
  asm volatile("global_load_async_to_lds_b128 %0, %1, off"
               :: "v"(lds_off), "v"(gaddr)
               : "memory");
}

// ---------------------------------------------------------------------------
// Generic WMMA GEMM: out[M,N] = act(A[M,K] @ W[N,K]^T + bias)
// 256 threads = 8 waves, one 16x16 output tile per wave.
// ---------------------------------------------------------------------------
__global__ void wmma_gemm_kernel(const _Float16* __restrict__ A, int lda,
                                 const _Float16* __restrict__ W,  // [N][K]
                                 const float* __restrict__ bias,  // nullable
                                 float* __restrict__ outF,        // nullable
                                 _Float16* __restrict__ outH,     // nullable
                                 int ldo, int Mtiles, int Ntiles, int K, int act) {
  int wave = (int)(threadIdx.x >> 5);
  long tile = (long)blockIdx.x * 8 + wave;
  long ntot = (long)Mtiles * Ntiles;
  if (tile >= ntot) return;  // wave-uniform: EXEC stays all-ones for WMMA
  int mt = (int)(tile / Ntiles);
  int nt = (int)(tile % Ntiles);
  long row0 = (long)mt * 16;
  int col0 = nt * 16;
  v8f acc = {};
  for (int k = 0; k < K; k += 32) {
    v16h a = frag_load(A + row0 * (long)lda + k, lda);
    v16h b = frag_load(W + (long)col0 * K + k, K);
    acc = __builtin_amdgcn_wmma_f32_16x16x32_f16(false, a, false, b, (short)0, acc,
                                                 false, false);
  }
  int lane = threadIdx.x & 31;
  int col = col0 + (lane & 15);
  long rbase = row0 + ((lane >> 4) << 3);
  float bv = bias ? bias[col] : 0.0f;
#pragma unroll
  for (int j = 0; j < 8; ++j) {
    float v = acc[j] + bv;
    if (act == 1) v = v > 0.0f ? v : 0.0f;
    long o = (rbase + j) * (long)ldo + col;
    if (outF) outF[o] = v;
    if (outH) outH[o] = (_Float16)v;
  }
}

// ---------------------------------------------------------------------------
// Prep: per (b,v) stats + x_comp forward-fill; writes f16 GRU/LSTM input rows.
// grid=B blocks of 32 threads (one wave, thread = feature v).
// ---------------------------------------------------------------------------
__global__ void prep_kernel(const float* __restrict__ values,
                            const float* __restrict__ masks,
                            const float* __restrict__ rainF,
                            const float* __restrict__ rainB,
                            const int* __restrict__ lengths,
                            float* __restrict__ stats,       // [B][97]
                            _Float16* __restrict__ gru_in,   // [BT][96]
                            _Float16* __restrict__ lstm_in)  // [BT][128]
{
  int b = blockIdx.x, v = threadIdx.x;
  int len = lengths[b];
  long base = (long)b * TSTEPS * VV + v;
  float ms = 0.f, sum = 0.f;
  for (int t = 0; t < TSTEPS; ++t) {
    long idx = base + (long)t * VV;
    float m = masks[idx];
    sum += m * values[idx];
    ms += m;
  }
  float mean = sum / fmaxf(ms, 1.0f);
  float dss = 0.f;
  float prev = values[base];  // x_prime[0] = values[0]
  for (int t = 0; t < TSTEPS; ++t) {
    long idx = base + (long)t * VV;
    float val = values[idx];
    float m = masks[idx];
    float d = val - mean;
    dss += d * d;
    float pm = (t < len) ? 1.0f : 0.0f;
    float xc = (m * val + (1.0f - m) * prev) * pm;
    prev = val;
    long row = (long)b * TSTEPS + t;
    gru_in[row * 96 + v]        = (_Float16)xc;
    gru_in[row * 96 + 32 + v]   = (_Float16)rainF[idx];
    gru_in[row * 96 + 64 + v]   = (_Float16)rainB[idx];
    lstm_in[row * 128 + v]      = (_Float16)xc;
    lstm_in[row * 128 + 32 + v] = (_Float16)m;
  }
  float var = (ms > 1.0f) ? dss / (ms - 1.0f) : 0.0f;
  stats[b * 97 + 1 + v]  = mean;
  stats[b * 97 + 33 + v] = sqrtf(var);
  stats[b * 97 + 65 + v] = 1.0f - ms / fmaxf((float)len, 1.0f);
  if (v == 0) stats[b * 97] = (float)len;
}

// Context MLP: stats(97)->relu(64)->32 ; <<<1,256>>>, thread = batch row
__global__ void cmlp_kernel(const float* __restrict__ stats,
                            const float* __restrict__ W1, const float* __restrict__ b1,
                            const float* __restrict__ W2, const float* __restrict__ b2,
                            float* __restrict__ ctx)  // [B][64], cols 0..31
{
  int b = threadIdx.x;
  float h1[64];
  for (int o = 0; o < 64; ++o) {
    float a = b1[o];
    for (int i = 0; i < 97; ++i) a += W1[o * 97 + i] * stats[b * 97 + i];
    h1[o] = a > 0.f ? a : 0.f;
  }
  for (int o = 0; o < 32; ++o) {
    float a = b2[o];
    for (int i = 0; i < 64; ++i) a += W2[o * 64 + i] * h1[i];
    ctx[b * 64 + o] = a;
  }
}

// ---------------------------------------------------------------------------
// Persistent GRU scan: <<<1,256,shmem>>>  thread = batch row; 8 waves do WMMA.
// gh = h @ Whh^T via v_wmma (16 Mtiles x 6 Ntiles = 96 tiles, 12/wave).
// gi rows are double-buffered in LDS via GLOBAL_LOAD_ASYNC_TO_LDS_B128,
// prefetched one step ahead and overlapped with the WMMA phase.
// Each thread prefetches only ITS OWN row, so a per-wave s_wait_asynccnt is a
// sufficient completion guarantee (no cross-wave dependency on gi data).
// LDS: hH 16KB + gh 96KB + gi double buffer 96KB = 208KB (< 320KB/WGP).
// ---------------------------------------------------------------------------
__global__ void gru_scan_kernel(const _Float16* __restrict__ gi,   // [BT][96] (+bih)
                                const _Float16* __restrict__ Whh,  // [96][32]
                                const float* __restrict__ bhh,     // [96]
                                const int* __restrict__ lengths,
                                float* __restrict__ ctx)           // [B][64] cols 32..63
{
  extern __shared__ char smem[];
  _Float16* hH  = (_Float16*)smem;                        // [256][32]   16384 B
  float*    gh  = (float*)(smem + 16384);                 // [256][96]   98304 B
  _Float16* giL = (_Float16*)(smem + 16384 + 98304);      // [2][256][96] 98304 B
  const unsigned giL_base = 16384u + 98304u;              // LDS byte offset of giL
  int tid = (int)threadIdx.x;
  int lane = tid & 31, wave = tid >> 5;
  float h[32], hn[32];
#pragma unroll
  for (int j = 0; j < 32; ++j) { h[j] = 0.f; hn[j] = 0.f; }
  for (int j = tid; j < 256 * 32; j += 256) hH[j] = (_Float16)0.f;
  int len = lengths[tid];
  // async prefetch of step 0's row into buffer 0 (12 x 16B per thread)
  {
    const _Float16* g0 = gi + (long)tid * TSTEPS * 96;
    unsigned l0 = giL_base + (unsigned)tid * 192u;
#pragma unroll
    for (int i = 0; i < 12; ++i) async_b128(l0 + i * 16u, g0 + i * 8);
  }
  __syncthreads();
  for (int t = 0; t < TSTEPS; ++t) {
    // issue async prefetch for step t+1 into the other buffer
    if (t + 1 < TSTEPS) {
      const _Float16* gn = gi + ((long)tid * TSTEPS + t + 1) * 96;
      unsigned ln = giL_base + (unsigned)(((t + 1) & 1) * 49152) + (unsigned)tid * 192u;
#pragma unroll
      for (int i = 0; i < 12; ++i) async_b128(ln + i * 16u, gn + i * 8);
    }
    // --- WMMA phase (overlaps with the in-flight async copies) ---
    for (int i = 0; i < 12; ++i) {
      int tile = wave * 12 + i;
      int mt = tile / 6, nt = tile % 6;
      v16h a = frag_load(hH + mt * 16 * 32, 32);
      v16h b = frag_load(Whh + nt * 16 * 32, 32);
      v8f acc = {};
      acc = __builtin_amdgcn_wmma_f32_16x16x32_f16(false, a, false, b, (short)0, acc,
                                                   false, false);
      int col = nt * 16 + (lane & 15);
      int rb = mt * 16 + ((lane >> 4) << 3);
#pragma unroll
      for (int j = 0; j < 8; ++j) gh[(rb + j) * 96 + col] = acc[j];
    }
    // retire step-t's async batch (leaves the t+1 prefetch in flight)
    if (t + 1 < TSTEPS) {
      asm volatile("s_wait_asynccnt 12" ::: "memory");
    } else {
      asm volatile("s_wait_asynccnt 0" ::: "memory");
    }
    __syncthreads();
    // --- gate phase: thread tid owns batch row tid ---
    const _Float16* gir = giL + (t & 1) * 24576 + tid * 96;
    float pm = (t < len) ? 1.0f : 0.0f;
#pragma unroll 4
    for (int j = 0; j < 32; ++j) {
      float r = sigf((float)gir[j]      + gh[tid * 96 + j]      + bhh[j]);
      float z = sigf((float)gir[32 + j] + gh[tid * 96 + 32 + j] + bhh[32 + j]);
      float n = tanhf((float)gir[64 + j] + r * (gh[tid * 96 + 64 + j] + bhh[64 + j]));
      float hh = (1.0f - z) * n + z * h[j];
      h[j] = hh;
      hn[j] = pm * hh + (1.0f - pm) * hn[j];
      hH[tid * 32 + j] = (_Float16)hh;
    }
    __syncthreads();
  }
  for (int j = 0; j < 32; ++j) ctx[tid * 64 + 32 + j] = hn[j];
}

// Broadcast context vec into lstm_in cols 64..127 (f16)
__global__ void ctxb_kernel(const float* __restrict__ ctx, _Float16* __restrict__ lstm_in) {
  long row = (long)blockIdx.x * 256 + threadIdx.x;
  if (row >= BTROWS) return;
  int b = (int)(row / TSTEPS);
  for (int j = 0; j < 64; ++j)
    lstm_in[row * 128 + 64 + j] = (_Float16)ctx[b * 64 + j];
}

// h0 = ctx @ init_W^T + b ; c0 = tanh(h0); also seed hiddens t=0 / t=T-1
__global__ void init_kernel(const float* __restrict__ ctx,
                            const float* __restrict__ iW, const float* __restrict__ ib,
                            float* __restrict__ h0, float* __restrict__ c0,
                            _Float16* __restrict__ hiddens)  // [BT][128]
{
  int b = threadIdx.x;  // 256
  for (int o = 0; o < 128; ++o) {
    float s = ib[o];
    for (int j = 0; j < 64; ++j) s += iW[o * 64 + j] * ctx[b * 64 + j];
    h0[b * 128 + o] = s;
    c0[b * 128 + o] = tanhf(s);
    long row = (o < 64) ? ((long)b * TSTEPS) : ((long)b * TSTEPS + TSTEPS - 1);
    hiddens[row * 128 + o] = (_Float16)s;
  }
}

// ---------------------------------------------------------------------------
// Persistent LSTM scan: grid=2 (fwd/bwd), 1024 threads, h in LDS f16.
// gh = h @ Whh^T : 16x16 tiles = 256 tiles, 32 waves -> 8 tiles/wave, K=64.
// Next step's gate rows are prefetched (global_prefetch_b8) during WMMA.
// ---------------------------------------------------------------------------
__global__ void lstm_scan_kernel(const _Float16* __restrict__ giF,
                                 const _Float16* __restrict__ giB,   // [BT][256] (+bih+bhh)
                                 const _Float16* __restrict__ WhhF,
                                 const _Float16* __restrict__ WhhB,  // [256][64]
                                 const float* __restrict__ h0,       // [B][128]
                                 const float* __restrict__ c0,       // [B][128]
                                 _Float16* __restrict__ hiddens)     // [BT][128]
{
  int dir = blockIdx.x;
  const _Float16* gi = dir ? giB : giF;
  const _Float16* Whh = dir ? WhhB : WhhF;
  extern __shared__ char smem[];
  _Float16* hH = (_Float16*)smem;             // [256][64]
  float* g = (float*)(smem + 256 * 64 * 2);   // [256][256]
  int tid = (int)threadIdx.x;
  int lane = tid & 31, wave = tid >> 5;
  int b = tid >> 2, j0 = (tid & 3) * 16;
  float c[16];
#pragma unroll
  for (int jj = 0; jj < 16; ++jj) {
    c[jj] = c0[b * 128 + dir * 64 + j0 + jj];
    hH[b * 64 + j0 + jj] = (_Float16)h0[b * 128 + dir * 64 + j0 + jj];
  }
  __syncthreads();
  for (int s = 0; s < TSTEPS - 1; ++s) {
    int t_in = dir ? (TSTEPS - 1 - s) : s;
    int t_st = dir ? (TSTEPS - 2 - s) : (s + 1);
    // prefetch next step's gate blocks toward the WGP while WMMA runs
    if (s + 1 < TSTEPS - 1) {
      int t_nx = dir ? (TSTEPS - 2 - s) : (s + 1);
      const _Float16* gnx = gi + ((long)b * TSTEPS + t_nx) * 256 + j0;
      __builtin_prefetch(gnx, 0, 3);
      __builtin_prefetch(gnx + 64, 0, 3);
      __builtin_prefetch(gnx + 128, 0, 3);
      __builtin_prefetch(gnx + 192, 0, 3);
    }
    // --- WMMA phase ---
    for (int i = 0; i < 8; ++i) {
      int tile = wave * 8 + i;
      int mt = tile >> 4, nt = tile & 15;
      v8f acc = {};
#pragma unroll
      for (int k = 0; k < 64; k += 32) {
        v16h a = frag_load(hH + mt * 16 * 64 + k, 64);
        v16h bb = frag_load(Whh + nt * 16 * 64 + k, 64);
        acc = __builtin_amdgcn_wmma_f32_16x16x32_f16(false, a, false, bb, (short)0, acc,
                                                     false, false);
      }
      int col = nt * 16 + (lane & 15);
      int rb = mt * 16 + ((lane >> 4) << 3);
#pragma unroll
      for (int j = 0; j < 8; ++j) g[(rb + j) * 256 + col] = acc[j];
    }
    __syncthreads();
    // --- gate phase: thread (b, j0..j0+15) ---
    const _Float16* gir = gi + ((long)b * TSTEPS + t_in) * 256;
#pragma unroll 4
    for (int jj = 0; jj < 16; ++jj) {
      int j = j0 + jj;
      float ig = sigf(g[b * 256 + j]        + (float)gir[j]);
      float fg = sigf(g[b * 256 + 64 + j]   + (float)gir[64 + j]);
      float gg = tanhf(g[b * 256 + 128 + j] + (float)gir[128 + j]);
      float og = sigf(g[b * 256 + 192 + j]  + (float)gir[192 + j]);
      c[jj] = fg * c[jj] + ig * gg;
      float hv = og * tanhf(c[jj]);
      hH[b * 64 + j] = (_Float16)hv;
      hiddens[((long)b * TSTEPS + t_st) * 128 + dir * 64 + j] = (_Float16)hv;
    }
    __syncthreads();
  }
}

// ---------------------------------------------------------------------------
// Small utility kernels
// ---------------------------------------------------------------------------
__global__ void f32tof16_kernel(const float* __restrict__ s, _Float16* __restrict__ d, int n) {
  int i = blockIdx.x * 256 + threadIdx.x;
  if (i < n) d[i] = (_Float16)s[i];
}
__global__ void addvec_kernel(const float* __restrict__ a, const float* __restrict__ b,
                              float* __restrict__ d, int n) {
  int i = blockIdx.x * 256 + threadIdx.x;
  if (i < n) d[i] = a[i] + b[i];
}
// feat_W (V,HF,V) -> f16 with column v==i zeroed; flat index identical
__global__ void featpack_kernel(const float* __restrict__ fw, _Float16* __restrict__ d) {
  int idx = blockIdx.x * 256 + threadIdx.x;  // 32768
  if (idx >= 32 * 32 * 32) return;
  int v = idx & 31;
  int i = idx >> 10;
  d[idx] = (_Float16)((v == i) ? 0.0f : fw[idx]);
}
// fuse: gamma==1, so fold fuse_W[:, :32] row-sums into the bias
__global__ void fusebias_kernel(const float* __restrict__ fW, const float* __restrict__ fb,
                                float* __restrict__ effb, _Float16* __restrict__ W2h) {
  int j = threadIdx.x;  // 32
  float s = fb[j];
  for (int k = 0; k < 32; ++k) s += fW[j * 64 + k];
  effb[j] = s;
  for (int k = 0; k < 32; ++k) W2h[j * 32 + k] = (_Float16)fW[j * 64 + 32 + k];
}
// feat_imp = relu(zh) @ nl2^T  (dot-32 per row) -- zh already relu'd by GEMM
__global__ void nl2_kernel(const _Float16* __restrict__ zh, const float* __restrict__ w,
                           const float* __restrict__ bb, float* __restrict__ out,
                           long rows, long rowOff) {
  long r = (long)blockIdx.x * 256 + threadIdx.x;
  if (r >= rows) return;
  float s = bb[0];
#pragma unroll
  for (int k = 0; k < 32; ++k) s += (float)zh[r * 32 + k] * w[k];
  out[rowOff + r] = s;
}
// final fuse + fill + padding
__global__ void final_kernel(const float* __restrict__ values, const float* __restrict__ masks,
                             const int* __restrict__ lengths,
                             const float* __restrict__ feat_imp,
                             const float* __restrict__ rnn_imp,
                             const float* __restrict__ fuse_logit,
                             float* __restrict__ out) {
  long idx = (long)blockIdx.x * 256 + threadIdx.x;
  if (idx >= (long)BTROWS * VV) return;
  long row = idx >> 5;
  int t = (int)(row % TSTEPS);
  int b = (int)(row / TSTEPS);
  float pm = (t < lengths[b]) ? 1.0f : 0.0f;
  float m = masks[idx], val = values[idx];
  float beta = sigf(fuse_logit[idx]);
  float imp = beta * feat_imp[idx] + (1.0f - beta) * rnn_imp[idx];
  out[idx] = pm * (m * val + (1.0f - m) * imp);
}

// ---------------------------------------------------------------------------
extern "C" void kernel_launch(void* const* d_in, const int* in_sizes, int n_in,
                              void* d_out, int out_size, void* d_ws, size_t ws_size,
                              hipStream_t stream) {
  (void)in_sizes; (void)n_in; (void)out_size; (void)ws_size;
  const float* values = (const float*)d_in[0];
  const float* masks  = (const float*)d_in[1];
  const float* rainF  = (const float*)d_in[2];
  const float* rainB  = (const float*)d_in[3];
  const float* cW1 = (const float*)d_in[4];  const float* cb1 = (const float*)d_in[5];
  const float* cW2 = (const float*)d_in[6];  const float* cb2 = (const float*)d_in[7];
  const float* gWih = (const float*)d_in[8]; const float* gWhh = (const float*)d_in[9];
  const float* gbih = (const float*)d_in[10]; const float* gbhh = (const float*)d_in[11];
  const float* iW = (const float*)d_in[12];  const float* ib = (const float*)d_in[13];
  const float* lfWih = (const float*)d_in[14]; const float* lfWhh = (const float*)d_in[15];
  const float* lfbih = (const float*)d_in[16]; const float* lfbhh = (const float*)d_in[17];
  const float* lbWih = (const float*)d_in[18]; const float* lbWhh = (const float*)d_in[19];
  const float* lbbih = (const float*)d_in[20]; const float* lbbhh = (const float*)d_in[21];
  const float* rW = (const float*)d_in[22];  const float* rb = (const float*)d_in[23];
  const float* featW = (const float*)d_in[24]; const float* featb = (const float*)d_in[25];
  const float* nl1W = (const float*)d_in[26];  const float* nl1b = (const float*)d_in[27];
  const float* nl2W = (const float*)d_in[28];  const float* nl2b = (const float*)d_in[29];
  const float* fuseW = (const float*)d_in[30]; const float* fuseb = (const float*)d_in[31];
  const int* lengths = (const int*)d_in[32];
  float* out = (float*)d_out;

  char* ws = (char*)d_ws;
  size_t off = 0;
  auto alloc = [&](size_t bytes) -> void* {
    void* p = ws + off;
    off = (off + bytes + 255) & ~(size_t)255;
    return p;
  };
  // f16 weights
  _Float16* gWih_h  = (_Float16*)alloc(96 * 96 * 2);
  _Float16* gWhh_h  = (_Float16*)alloc(96 * 32 * 2);
  _Float16* lfWih_h = (_Float16*)alloc(256 * 128 * 2);
  _Float16* lbWih_h = (_Float16*)alloc(256 * 128 * 2);
  _Float16* lfWhh_h = (_Float16*)alloc(256 * 64 * 2);
  _Float16* lbWhh_h = (_Float16*)alloc(256 * 64 * 2);
  _Float16* rW_h    = (_Float16*)alloc(32 * 128 * 2);
  _Float16* featp_h = (_Float16*)alloc(1024 * 32 * 2);
  _Float16* nl1_h   = (_Float16*)alloc(32 * 32 * 2);
  _Float16* fuse2_h = (_Float16*)alloc(32 * 32 * 2);
  // f32 small
  float* stats  = (float*)alloc(256 * 97 * 4);
  float* ctx    = (float*)alloc(256 * 64 * 4);
  float* h0     = (float*)alloc(256 * 128 * 4);
  float* c0     = (float*)alloc(256 * 128 * 4);
  float* biasF  = (float*)alloc(256 * 4);
  float* biasB  = (float*)alloc(256 * 4);
  float* effb   = (float*)alloc(32 * 4);
  // big buffers
  _Float16* gru_in  = (_Float16*)alloc((size_t)BTROWS * 96 * 2);   // aliased later: hid_chunk
  _Float16* gi_gru  = (_Float16*)alloc((size_t)BTROWS * 96 * 2);   // aliased later: zh_chunk
  _Float16* lstm_in = (_Float16*)alloc((size_t)BTROWS * 128 * 2);
  _Float16* gi_f    = (_Float16*)alloc((size_t)BTROWS * 256 * 2);
  _Float16* gi_b    = (_Float16*)alloc((size_t)BTROWS * 256 * 2);
  _Float16* hiddens = (_Float16*)alloc((size_t)BTROWS * 128 * 2);
  float* rnn_imp    = (float*)alloc((size_t)BTROWS * 32 * 4);
  float* fuse_lg    = (float*)alloc((size_t)BTROWS * 32 * 4);
  float* feat_imp   = (float*)alloc((size_t)BTROWS * 32 * 4);

  // --- weight prep ---
  f32tof16_kernel<<<(96 * 96 + 255) / 256, 256, 0, stream>>>(gWih, gWih_h, 96 * 96);
  f32tof16_kernel<<<(96 * 32 + 255) / 256, 256, 0, stream>>>(gWhh, gWhh_h, 96 * 32);
  f32tof16_kernel<<<128, 256, 0, stream>>>(lfWih, lfWih_h, 256 * 128);
  f32tof16_kernel<<<128, 256, 0, stream>>>(lbWih, lbWih_h, 256 * 128);
  f32tof16_kernel<<<64, 256, 0, stream>>>(lfWhh, lfWhh_h, 256 * 64);
  f32tof16_kernel<<<64, 256, 0, stream>>>(lbWhh, lbWhh_h, 256 * 64);
  f32tof16_kernel<<<16, 256, 0, stream>>>(rW, rW_h, 32 * 128);
  f32tof16_kernel<<<4, 256, 0, stream>>>(nl1W, nl1_h, 32 * 32);
  featpack_kernel<<<128, 256, 0, stream>>>(featW, featp_h);
  addvec_kernel<<<1, 256, 0, stream>>>(lfbih, lfbhh, biasF, 256);
  addvec_kernel<<<1, 256, 0, stream>>>(lbbih, lbbhh, biasB, 256);
  fusebias_kernel<<<1, 32, 0, stream>>>(fuseW, fuseb, effb, fuse2_h);

  // --- stats + inputs ---
  prep_kernel<<<BB, 32, 0, stream>>>(values, masks, rainF, rainB, lengths,
                                     stats, gru_in, lstm_in);
  cmlp_kernel<<<1, 256, 0, stream>>>(stats, cW1, cb1, cW2, cb2, ctx);

  // --- GRU input projection + scan ---
  {
    int Mt = BTROWS / 16, Nt = 6;  // N=96
    wmma_gemm_kernel<<<(Mt * Nt + 7) / 8, 256, 0, stream>>>(
        gru_in, 96, gWih_h, gbih, nullptr, gi_gru, 96, Mt, Nt, 96, 0);
  }
  {
    size_t sh = 16384 + 98304 + 98304;  // hH + gh + gi double buffer = 208 KB
    gru_scan_kernel<<<1, 256, sh, stream>>>(gi_gru, gWhh_h, gbhh, lengths, ctx);
  }

  ctxb_kernel<<<(BTROWS + 255) / 256, 256, 0, stream>>>(ctx, lstm_in);
  init_kernel<<<1, 256, 0, stream>>>(ctx, iW, ib, h0, c0, hiddens);

  // --- LSTM input projections (the big hoisted GEMMs) + scans ---
  {
    int Mt = BTROWS / 16, Nt = 16;  // N=256, K=128
    wmma_gemm_kernel<<<(Mt * Nt + 7) / 8, 256, 0, stream>>>(
        lstm_in, 128, lfWih_h, biasF, nullptr, gi_f, 256, Mt, Nt, 128, 0);
    wmma_gemm_kernel<<<(Mt * Nt + 7) / 8, 256, 0, stream>>>(
        lstm_in, 128, lbWih_h, biasB, nullptr, gi_b, 256, Mt, Nt, 128, 0);
  }
  {
    size_t sh = 256 * 64 * 2 + 256 * 256 * 4;  // 288 KB of the 320 KB WGP LDS
    lstm_scan_kernel<<<2, 1024, sh, stream>>>(gi_f, gi_b, lfWhh_h, lbWhh_h, h0, c0,
                                              hiddens);
  }

  // --- rnn_imp = hiddens @ rimp_W^T + b ---
  {
    int Mt = BTROWS / 16, Nt = 2;  // N=32, K=128
    wmma_gemm_kernel<<<(Mt * Nt + 7) / 8, 256, 0, stream>>>(
        hiddens, 128, rW_h, rb, rnn_imp, nullptr, 32, Mt, Nt, 128, 0);
  }
  // --- fuse logits = masks @ Wf2^T + (Wf1@1 + b) ---
  {
    int Mt = BTROWS / 16, Nt = 2;  // K=32; A = masks columns of lstm_in
    wmma_gemm_kernel<<<(Mt * Nt + 7) / 8, 256, 0, stream>>>(
        lstm_in + 32, 128, fuse2_h, effb, fuse_lg, nullptr, 32, Mt, Nt, 32, 0);
  }

  // --- feature branch, chunked to reuse dead GRU buffers ---
  {
    const int NC = 16;
    const long CHR = BTROWS / NC;  // 8192 bt-rows per chunk
    _Float16* hid_chunk = gru_in;  // 8192*1024 f16 = 16.8 MB (fits 25 MB region)
    _Float16* zh_chunk  = gi_gru;  // 262144*32 f16 = 16.8 MB
    for (int cc = 0; cc < NC; ++cc) {
      long r0 = (long)cc * CHR;
      int Mt1 = (int)(CHR / 16), Nt1 = 64;  // N=1024, K=32
      wmma_gemm_kernel<<<(Mt1 * Nt1 + 7) / 8, 256, 0, stream>>>(
          lstm_in + r0 * 128, 128, featp_h, featb, nullptr, hid_chunk, 1024,
          Mt1, Nt1, 32, 1);
      int Mt2 = (int)(CHR * 32 / 16), Nt2 = 2;  // rows' = CHR*32, N=32, K=32
      wmma_gemm_kernel<<<(Mt2 * Nt2 + 7) / 8, 256, 0, stream>>>(
          hid_chunk, 32, nl1_h, nl1b, nullptr, zh_chunk, 32, Mt2, Nt2, 32, 1);
      long rows = CHR * 32;
      nl2_kernel<<<(int)((rows + 255) / 256), 256, 0, stream>>>(
          zh_chunk, nl2W, nl2b, feat_imp, rows, r0 * 32);
    }
  }

  // --- final fuse + fill + padding mask ---
  {
    long total = (long)BTROWS * VV;
    final_kernel<<<(int)((total + 255) / 256), 256, 0, stream>>>(
        values, masks, lengths, feat_imp, rnn_imp, fuse_lg, out);
  }
}
